// OrientedTextPostProcessing_13314398617721
// MI455X (gfx1250) — compile-verified, matrix operand order checked by reading, not run on hardware
//
#include <hip/hip_runtime.h>
#include <stdint.h>

typedef __attribute__((ext_vector_type(16))) _Float16 v16h;
typedef __attribute__((ext_vector_type(8)))  float    v8f;

static constexpr int Hc = 512, Wc = 512, HW = Hc * Wc;
static constexpr int KW = 1024, KC = 4096, NUM_CLS = 68;
static constexpr float SCORE_MIN_WORD = 0.5f;
static constexpr float SCORE_MIN_CHAR = 0.25f;
static constexpr float WORD_NMS_THR = 0.5f, CHAR_NMS_THR = 0.3f;
static constexpr float STRIDE = 4.0f;

// ---------------- IoU on AABBs (shared by colmax + assign so bits match) ----
__device__ __forceinline__ float iou4(float ax1, float ay1, float ax2, float ay2,
                                      float bx1, float by1, float bx2, float by2) {
  float ix = fmaxf(0.0f, fminf(ax2, bx2) - fmaxf(ax1, bx1));
  float iy = fmaxf(0.0f, fminf(ay2, by2) - fmaxf(ay1, by1));
  float inter = ix * iy;
  float aa = (ax2 - ax1) * (ay2 - ay1);
  float ab = (bx2 - bx1) * (by2 - by1);
  return inter / fmaxf(aa + ab - inter, 1e-6f);
}

// ---------------- top-k: key init ------------------------------------------
__global__ void k_init_keys_word(const float* __restrict__ wfg,
                                 unsigned long long* __restrict__ keys) {
  int i = blockIdx.x * blockDim.x + threadIdx.x;
  if (i >= HW) return;
  unsigned bits = __float_as_uint(wfg[i]);               // scores >= 0
  keys[i] = ((unsigned long long)bits << 32) | (unsigned)(~i);
}

__global__ void k_init_keys_char(const float* __restrict__ wfg,
                                 const float* __restrict__ cfg,
                                 unsigned long long* __restrict__ keys) {
  int i = blockIdx.x * blockDim.x + threadIdx.x;
  if (i >= HW) return;
  float s = (wfg[i] > SCORE_MIN_WORD && cfg[i] > SCORE_MIN_CHAR) ? cfg[i] : 0.0f;
  unsigned bits = __float_as_uint(s);
  keys[i] = ((unsigned long long)bits << 32) | (unsigned)(~i);
}

// ---------------- bitonic sort (descending), global substage ----------------
__global__ void k_bitonic_global(unsigned long long* __restrict__ keys,
                                 unsigned j, unsigned k) {
  unsigned i = blockIdx.x * blockDim.x + threadIdx.x;
  unsigned ixj = i ^ j;
  if (ixj > i && i < (unsigned)HW) {
    unsigned long long a = keys[i], b = keys[ixj];
    bool desc = ((i & k) == 0);
    if (desc ? (a < b) : (a > b)) { keys[i] = b; keys[ixj] = a; }
  }
}

// fused LDS substages: handles all j from jstart (<=1024) down to 1
__global__ void k_bitonic_local(unsigned long long* __restrict__ keys,
                                unsigned k, unsigned jstart) {
  __shared__ unsigned long long s[2048];
  unsigned base = blockIdx.x * 2048u;
  for (unsigned t = threadIdx.x; t < 2048u; t += blockDim.x) s[t] = keys[base + t];
  __syncthreads();
  for (unsigned j = jstart; j > 0; j >>= 1) {
    for (unsigned t = threadIdx.x; t < 1024u; t += blockDim.x) {
      unsigned i = ((t & ~(j - 1u)) << 1) | (t & (j - 1u));
      unsigned long long a = s[i], b = s[i + j];
      bool desc = (((base + i) & k) == 0);
      if (desc ? (a < b) : (a > b)) { s[i] = b; s[i + j] = a; }
    }
    __syncthreads();
  }
  for (unsigned t = threadIdx.x; t < 2048u; t += blockDim.x) keys[base + t] = s[t];
}

__global__ void k_extract(const unsigned long long* __restrict__ keys,
                          int* __restrict__ idx, float* __restrict__ score, int k) {
  int r = blockIdx.x * blockDim.x + threadIdx.x;
  if (r >= k) return;
  unsigned long long key = keys[r];
  idx[r] = (int)(~(unsigned)(key & 0xffffffffull));
  score[r] = __uint_as_float((unsigned)(key >> 32));
}

// ---------------- decode selected boxes ------------------------------------
__global__ void k_decode(const float* __restrict__ tblr, const float* __restrict__ orient,
                         const int* __restrict__ sel, const int* __restrict__ p_sw,
                         const int* __restrict__ p_sh, int k,
                         float* __restrict__ boxes8, float* __restrict__ aabb) {
  int r = blockIdx.x * blockDim.x + threadIdx.x;
  if (r >= k) return;
  int idx = sel[r];
  int y = idx / Wc, x = idx - y * Wc;
  float t  = tblr[0 * HW + idx], b = tblr[1 * HW + idx];
  float l  = tblr[2 * HW + idx], rr = tblr[3 * HW + idx];
  float o  = orient ? orient[idx] : 0.0f;
  float fx = (float)(*p_sw) * STRIDE, fy = (float)(*p_sh) * STRIDE;
  float xf = (float)x, yf = (float)y;
  float xmin = fx * (xf - l), xmax = fx * (xf + rr);
  float ymin = fy * (yf - t), ymax = fy * (yf + b);
  float cx = fx * xf, cy = fy * yf;
  float co = cosf(o), so = sinf(o);
  float bx[8];
  {
    float dx, dy;
    dx = xmin - cx; dy = ymin - cy; bx[0] = cx + dx * co - dy * so; bx[1] = cy + dx * so + dy * co;
    dx = xmax - cx; dy = ymin - cy; bx[2] = cx + dx * co - dy * so; bx[3] = cy + dx * so + dy * co;
    dx = xmax - cx; dy = ymax - cy; bx[4] = cx + dx * co - dy * so; bx[5] = cy + dx * so + dy * co;
    dx = xmin - cx; dy = ymax - cy; bx[6] = cx + dx * co - dy * so; bx[7] = cy + dx * so + dy * co;
  }
#pragma unroll
  for (int i2 = 0; i2 < 8; i2++) boxes8[r * 8 + i2] = bx[i2];
  aabb[r * 4 + 0] = fminf(fminf(bx[0], bx[2]), fminf(bx[4], bx[6]));
  aabb[r * 4 + 1] = fminf(fminf(bx[1], bx[3]), fminf(bx[5], bx[7]));
  aabb[r * 4 + 2] = fmaxf(fmaxf(bx[0], bx[2]), fmaxf(bx[4], bx[6]));
  aabb[r * 4 + 3] = fmaxf(fmaxf(bx[1], bx[3]), fmaxf(bx[5], bx[7]));
}

// ---------------- fast NMS: sup[j] = max_{i<j} IoU -------------------------
__global__ void k_nms(const float* __restrict__ aabb, const float* __restrict__ score,
                      int* __restrict__ keep, int k, float thr, float mins) {
  int j = blockIdx.x * blockDim.x + threadIdx.x;
  if (j >= k) return;
  float bx1 = aabb[j * 4 + 0], by1 = aabb[j * 4 + 1];
  float bx2 = aabb[j * 4 + 2], by2 = aabb[j * 4 + 3];
  float sup = 0.0f;
  for (int i = 0; i < j; i++) {
    sup = fmaxf(sup, iou4(aabb[i * 4 + 0], aabb[i * 4 + 1], aabb[i * 4 + 2], aabb[i * 4 + 3],
                          bx1, by1, bx2, by2));
  }
  keep[j] = (sup <= thr) && (score[j] > mins);
}

// ---------------- clip+round, emit 9-wide rows + clipped AABBs --------------
__global__ void k_clip_round(const float* __restrict__ b8, const float* __restrict__ score,
                             const int* __restrict__ keep, const int* __restrict__ p_w,
                             const int* __restrict__ p_h, float* __restrict__ out9,
                             float* __restrict__ aabb8, int k) {
  int j = blockIdx.x * blockDim.x + threadIdx.x;
  if (j >= k) return;
  float wmax = (float)(*p_w) - 1.0f, hmax = (float)(*p_h) - 1.0f;
  float v[8];
#pragma unroll
  for (int i = 0; i < 8; i++) {
    float t = rintf(b8[j * 8 + i]);                       // round half-to-even
    float lim = (i & 1) ? hmax : wmax;
    v[i] = fminf(fmaxf(t, 0.0f), lim);
    out9[j * 9 + i] = v[i];
  }
  out9[j * 9 + 8] = score[j] * (keep[j] ? 1.0f : 0.0f);
  aabb8[j * 4 + 0] = fminf(fminf(v[0], v[2]), fminf(v[4], v[6]));
  aabb8[j * 4 + 1] = fminf(fminf(v[1], v[3]), fminf(v[5], v[7]));
  aabb8[j * 4 + 2] = fmaxf(fmaxf(v[0], v[2]), fmaxf(v[4], v[6]));
  aabb8[j * 4 + 3] = fmaxf(fmaxf(v[1], v[3]), fmaxf(v[5], v[7]));
}

// ---------------- gather char class scores + per-char max conf --------------
__global__ void k_char_cls(const float* __restrict__ ccls, const int* __restrict__ cidx,
                           const int* __restrict__ ckeep, float* __restrict__ out_cls,
                           float* __restrict__ conf) {
  int j = blockIdx.x * blockDim.x + threadIdx.x;
  if (j >= KC) return;
  int idx = cidx[j];
  float kp = ckeep[j] ? 1.0f : 0.0f;
  float mx = -1e30f;
  for (int c = 0; c < NUM_CLS; c++) {
    float v = ccls[c * HW + idx] * kp;
    out_cls[j * NUM_CLS + c] = v;
    mx = fmaxf(mx, v);
  }
  conf[j] = mx;
}

// ---------------- column max of the 1024x4096 word-char IoU matrix ----------
__global__ void k_colmax(const float* __restrict__ wa, const float* __restrict__ ca,
                         float* __restrict__ colmax) {
  int j = blockIdx.x * blockDim.x + threadIdx.x;
  if (j >= KC) return;
  float bx1 = ca[j * 4 + 0], by1 = ca[j * 4 + 1], bx2 = ca[j * 4 + 2], by2 = ca[j * 4 + 3];
  float m = -1e30f;
  for (int i = 0; i < KW; i++) {
    m = fmaxf(m, iou4(wa[i * 4 + 0], wa[i * 4 + 1], wa[i * 4 + 2], wa[i * 4 + 3],
                      bx1, by1, bx2, by2));
  }
  colmax[j] = m;
}

// ---------------- text_score via WMMA: [num,cnt] = assign x [conf, 1] -------
// One wave per 16-word tile. A = 16x32 f16 assign tile, B col0=conf, col1=1.
// Fragment fill is fully branchless (float masks + selects) so the inner loop
// is straight-line VALU + one v_wmma per 32-char chunk.
__global__ void k_textscore(const float* __restrict__ waabb8, const float* __restrict__ caabb8,
                            const int* __restrict__ wkeep, const int* __restrict__ ckeep,
                            const float* __restrict__ colmax, const float* __restrict__ conf,
                            float* __restrict__ out_ts) {
  __shared__ float s_ca[32 * 4];
  __shared__ float s_cf[32];
  __shared__ float s_cm[32];
  __shared__ float s_ckf[32];
  __shared__ float s_num[16], s_cnt[16];

  const int tile = blockIdx.x;       // 0..63
  const int ln = threadIdx.x;        // 0..31 (one wave32)
  const int mrow = tile * 16 + (ln & 15);
  const int hi = ln >> 4;            // K-half selector for 16-bit A/B layout
  const int ncol = ln & 15;          // N column this lane feeds in B

  const float wx1 = waabb8[mrow * 4 + 0], wy1 = waabb8[mrow * 4 + 1];
  const float wx2 = waabb8[mrow * 4 + 2], wy2 = waabb8[mrow * 4 + 3];
  const float wkf = wkeep[mrow] ? 1.0f : 0.0f;
  const float bsel0 = (ncol == 0) ? 1.0f : 0.0f;   // lane feeds conf column
  const float bsel1 = (ncol == 1) ? 1.0f : 0.0f;   // lane feeds ones column

  v8f acc = {};
  for (int cb = 0; cb < KC; cb += 32) {
    __syncthreads();
    {
      int j = cb + ln;
      s_ca[ln * 4 + 0] = caabb8[j * 4 + 0];
      s_ca[ln * 4 + 1] = caabb8[j * 4 + 1];
      s_ca[ln * 4 + 2] = caabb8[j * 4 + 2];
      s_ca[ln * 4 + 3] = caabb8[j * 4 + 3];
      s_cf[ln] = conf[j];
      s_cm[ln] = colmax[j];
      s_ckf[ln] = ckeep[j] ? 1.0f : 0.0f;
    }
    __syncthreads();

    v16h a, b;
#pragma unroll
    for (int i = 0; i < 16; i++) {
      // 16-bit A 16x32 layout: lanes0-15 K=0..7 (V0..3) & 16..23 (V4..7); lanes16-31 +8
      int kk = (i < 8) ? (hi * 8 + i) : (16 + hi * 8 + (i - 8));
      float iou = iou4(wx1, wy1, wx2, wy2,
                       s_ca[kk * 4 + 0], s_ca[kk * 4 + 1], s_ca[kk * 4 + 2], s_ca[kk * 4 + 3]);
      float hit = ((iou > 0.0f) & (iou == s_cm[kk])) ? 1.0f : 0.0f;   // v_cndmask
      float av = wkf * s_ckf[kk] * hit;
      a[i] = (_Float16)av;
      float bv = bsel0 * s_cf[kk] + bsel1;                            // fma
      b[i] = (_Float16)bv;
    }
    acc = __builtin_amdgcn_wmma_f32_16x16x32_f16(false, a, false, b, (short)0, acc,
                                                 false, false);
  }

  // D layout: lane(0-15)=N, VGPR v -> M=v ; lane(16-31)=N, VGPR v -> M=8+v
  if (ln == 0)  { for (int v = 0; v < 8; v++) s_num[v]     = acc[v]; }
  if (ln == 16) { for (int v = 0; v < 8; v++) s_num[8 + v] = acc[v]; }
  if (ln == 1)  { for (int v = 0; v < 8; v++) s_cnt[v]     = acc[v]; }
  if (ln == 17) { for (int v = 0; v < 8; v++) s_cnt[8 + v] = acc[v]; }
  __syncthreads();
  if (ln < 16) out_ts[tile * 16 + ln] = s_num[ln] / fmaxf(s_cnt[ln], 1.0f);
}

// ---------------------------------------------------------------------------
extern "C" void kernel_launch(void* const* d_in, const int* in_sizes, int n_in,
                              void* d_out, int out_size, void* d_ws, size_t ws_size,
                              hipStream_t stream) {
  const float* wfg    = (const float*)d_in[0];
  const float* wtblr  = (const float*)d_in[1];
  const float* worient= (const float*)d_in[2];
  const float* cfg    = (const float*)d_in[3];
  const float* ctblr  = (const float*)d_in[4];
  const float* ccls   = (const float*)d_in[5];
  const int*   p_sw   = (const int*)d_in[6];
  const int*   p_sh   = (const int*)d_in[7];
  const int*   p_w    = (const int*)d_in[8];
  const int*   p_h    = (const int*)d_in[9];

  char* ws = (char*)d_ws;
  size_t off = 0;
  auto take = [&](size_t bytes) -> void* {
    void* p = ws + off;
    off += (bytes + 255) & ~(size_t)255;
    return p;
  };
  unsigned long long* keys = (unsigned long long*)take((size_t)HW * 8);
  int*   widx   = (int*)  take(KW * 4);
  float* wscore = (float*)take(KW * 4);
  int*   cidx   = (int*)  take(KC * 4);
  float* cscore = (float*)take(KC * 4);
  float* wb_raw = (float*)take(KW * 8 * 4);
  float* waabb  = (float*)take(KW * 4 * 4);
  float* waabb8 = (float*)take(KW * 4 * 4);
  int*   wkeep  = (int*)  take(KW * 4);
  float* cb_raw = (float*)take(KC * 8 * 4);
  float* caabb  = (float*)take(KC * 4 * 4);
  float* caabb8 = (float*)take(KC * 4 * 4);
  int*   ckeep  = (int*)  take(KC * 4);
  float* colmax = (float*)take(KC * 4);
  float* conf   = (float*)take(KC * 4);

  float* out_word = (float*)d_out;                 // 1024 x 9
  float* out_char = out_word + KW * 9;             // 4096 x 9
  float* out_cls  = out_char + KC * 9;             // 4096 x 68
  float* out_ts   = out_cls + KC * NUM_CLS;        // 1024

  auto sort_desc = [&](unsigned long long* kk) {
    for (unsigned k = 2; k <= (unsigned)HW; k <<= 1) {
      unsigned j = k >> 1;
      for (; j > 1024; j >>= 1)
        k_bitonic_global<<<HW / 256, 256, 0, stream>>>(kk, j, k);
      k_bitonic_local<<<HW / 2048, 256, 0, stream>>>(kk, k, j);
    }
  };

  // ---- word branch ----
  k_init_keys_word<<<HW / 256, 256, 0, stream>>>(wfg, keys);
  sort_desc(keys);
  k_extract<<<(KW + 255) / 256, 256, 0, stream>>>(keys, widx, wscore, KW);
  k_decode<<<(KW + 255) / 256, 256, 0, stream>>>(wtblr, worient, widx, p_sw, p_sh, KW,
                                                 wb_raw, waabb);
  k_nms<<<(KW + 255) / 256, 256, 0, stream>>>(waabb, wscore, wkeep, KW,
                                              WORD_NMS_THR, SCORE_MIN_WORD);
  k_clip_round<<<(KW + 255) / 256, 256, 0, stream>>>(wb_raw, wscore, wkeep, p_w, p_h,
                                                     out_word, waabb8, KW);

  // ---- char branch ----
  k_init_keys_char<<<HW / 256, 256, 0, stream>>>(wfg, cfg, keys);
  sort_desc(keys);
  k_extract<<<(KC + 255) / 256, 256, 0, stream>>>(keys, cidx, cscore, KC);
  k_decode<<<(KC + 255) / 256, 256, 0, stream>>>(ctblr, (const float*)nullptr, cidx,
                                                 p_sw, p_sh, KC, cb_raw, caabb);
  k_nms<<<(KC + 255) / 256, 256, 0, stream>>>(caabb, cscore, ckeep, KC,
                                              CHAR_NMS_THR, SCORE_MIN_CHAR);
  k_clip_round<<<(KC + 255) / 256, 256, 0, stream>>>(cb_raw, cscore, ckeep, p_w, p_h,
                                                     out_char, caabb8, KC);
  k_char_cls<<<(KC + 255) / 256, 256, 0, stream>>>(ccls, cidx, ckeep, out_cls, conf);

  // ---- assignment + text score ----
  k_colmax<<<(KC + 255) / 256, 256, 0, stream>>>(waabb8, caabb8, colmax);
  k_textscore<<<KW / 16, 32, 0, stream>>>(waabb8, caabb8, wkeep, ckeep, colmax, conf,
                                          out_ts);
}
